// CausalSelfAttention_68281390071867
// MI455X (gfx1250) — matrix operand order repaired and checked by reading, not compiled, not run on hardware
//
#include <hip/hip_runtime.h>
#include <cstdint>
#include <cstddef>

// ---------------------------------------------------------------------------
// CDNA5 (gfx1250) causal self-attention block, bf16 WMMA everywhere.
//   B=2, T=2048, C=2048, heads=32, kv-heads=8, hs=64, qkv width=3072
// GEMMs: async global->LDS DMA (ASYNCcnt) + double-buffered LDS when the
// gfx1250 async builtins are available; synchronous staging otherwise.
// ---------------------------------------------------------------------------

typedef __attribute__((ext_vector_type(16))) __bf16 v16bf;
typedef __attribute__((ext_vector_type(8)))  float  v8f;

union FragU { uint4 u[2]; v16bf v; };   // 32 bytes = 16 bf16 = one WMMA operand

#if __has_builtin(__builtin_amdgcn_global_load_async_to_lds_b128) && \
    __has_builtin(__builtin_amdgcn_s_wait_asynccnt)
#define USE_ASYNC_COPY 1
#else
#define USE_ASYNC_COPY 0
#endif

#if USE_ASYNC_COPY
// Builtin signature is (v4i addrspace(1)*, v4i addrspace(3)*, imm off, imm cpol)
typedef int v4i_gcc __attribute__((vector_size(16)));
typedef __attribute__((address_space(1))) v4i_gcc* g_v4i_ptr;
typedef __attribute__((address_space(3))) v4i_gcc* l_v4i_ptr;
// Per-lane 16B DMA: LDS[lptr] = MEM[gptr], tracked by ASYNCcnt.
#define ASYNC_B128(gptr, lptr)                                              \
  __builtin_amdgcn_global_load_async_to_lds_b128(                           \
      (g_v4i_ptr)(uintptr_t)(gptr), (l_v4i_ptr)(lptr), 0, 0)
#endif

__device__ __forceinline__ unsigned short f32_to_bf16(float f) {
  unsigned int u = __float_as_uint(f);
  u += 0x7FFFu + ((u >> 16) & 1u);      // round-to-nearest-even
  return (unsigned short)(u >> 16);
}

__device__ __forceinline__ v8f wmma_bf16(const FragU& a, const FragU& b, v8f c) {
  // D = A(16x32 bf16) x B(32x16 bf16) + C(16x16 f32)
  return __builtin_amdgcn_wmma_f32_16x16x32_bf16(
      false, a.v, false, b.v, (short)0, c, false, false);
}

// ---------------------------------------------------------------------------
// Elementwise f32 -> bf16 cast
// ---------------------------------------------------------------------------
__global__ void k_cast_bf16(const float* __restrict__ in,
                            unsigned short* __restrict__ out, int n) {
  int i = blockIdx.x * blockDim.x + threadIdx.x;
  if (i < n) out[i] = f32_to_bf16(in[i]);
}

// ---------------------------------------------------------------------------
// C[M,N] = A[M,K] * W[N,K]^T + bias[N]   (A,W bf16 row-major, C f32)
// Block tile 128x128, BK=32, double-buffered LDS. 8 waves: 2(M) x 4(N);
// wave tile 64x32 = 8 WMMA tiles.
// ---------------------------------------------------------------------------
__global__ __launch_bounds__(256) void k_gemm_bf16(
    const unsigned short* __restrict__ A,
    const unsigned short* __restrict__ W,
    const float* __restrict__ bias,
    float* __restrict__ C,
    int M, int N, int K) {
  __shared__ unsigned short As[2][128 * 32];
  __shared__ unsigned short Bs[2][128 * 32];

  const int tid   = threadIdx.x;
  const int lane  = tid & 31;
  const int wid   = tid >> 5;
  const int waveM = wid >> 2;           // 0..1
  const int waveN = wid & 3;            // 0..3
  const int m0 = blockIdx.y * 128;
  const int n0 = blockIdx.x * 128;
  const int lrow = lane & 15;
  const int half = lane >> 4;
  const int off8 = half * 8;            // A-frag chunk start within 32-K block
  const int koff = half * 8;            // B-frag chunk start within 32-K block

  const v8f vzero = {0.f, 0.f, 0.f, 0.f, 0.f, 0.f, 0.f, 0.f};
  v8f acc[4][2];
#pragma unroll
  for (int i = 0; i < 4; ++i)
#pragma unroll
    for (int j = 0; j < 2; ++j) acc[i][j] = vzero;

  // Tile-fill assignment: thread -> (row 0..127, k-chunk 0/16); 2x16B each buf.
  const int fr = tid >> 1;
  const int fk = (tid & 1) * 16;
  const unsigned short* gA = A + (size_t)(m0 + fr) * K + fk;
  const unsigned short* gW = W + (size_t)(n0 + fr) * K + fk;
  unsigned short* sA[2] = {&As[0][fr * 32 + fk], &As[1][fr * 32 + fk]};
  unsigned short* sB[2] = {&Bs[0][fr * 32 + fk], &Bs[1][fr * 32 + fk]};

  const int nk = K >> 5;

#if USE_ASYNC_COPY
  // Prologue: DMA tile 0 into buffer 0.
  ASYNC_B128(gA,     sA[0]);
  ASYNC_B128(gA + 8, sA[0] + 8);
  ASYNC_B128(gW,     sB[0]);
  ASYNC_B128(gW + 8, sB[0] + 8);
#endif

  for (int ik = 0; ik < nk; ++ik) {
    const int cur = ik & 1;
#if USE_ASYNC_COPY
    __builtin_amdgcn_s_wait_asynccnt(0);   // my tile-ik DMAs landed
    __syncthreads();                        // everyone's landed; buf[cur^1] free
    if (ik + 1 < nk) {
      const unsigned short* ga = gA + (size_t)(ik + 1) * 32;
      const unsigned short* gw = gW + (size_t)(ik + 1) * 32;
      ASYNC_B128(ga,     sA[cur ^ 1]);
      ASYNC_B128(ga + 8, sA[cur ^ 1] + 8);
      ASYNC_B128(gw,     sB[cur ^ 1]);
      ASYNC_B128(gw + 8, sB[cur ^ 1] + 8);
      __builtin_prefetch(ga + 32, 0, 0);    // tile ik+2 -> global_prefetch_b8
      __builtin_prefetch(gw + 32, 0, 0);
    }
#else
    {
      const uint4* ga4 = (const uint4*)(gA + (size_t)ik * 32);
      const uint4* gw4 = (const uint4*)(gW + (size_t)ik * 32);
      uint4 a0 = ga4[0], a1 = ga4[1];
      uint4 w0 = gw4[0], w1 = gw4[1];
      *(uint4*)sA[cur]       = a0;
      *(uint4*)(sA[cur] + 8) = a1;
      *(uint4*)sB[cur]       = w0;
      *(uint4*)(sB[cur] + 8) = w1;
      __syncthreads();
    }
#endif

    const unsigned short* as = As[cur];
    const unsigned short* bs = Bs[cur];
    FragU af[4], bf2[2];
#pragma unroll
    for (int i = 0; i < 4; ++i) {
      const unsigned short* p = &as[(waveM * 64 + i * 16 + lrow) * 32];
      af[i].u[0] = *(const uint4*)(p + off8);
      af[i].u[1] = *(const uint4*)(p + off8 + 16);
    }
#pragma unroll
    for (int j = 0; j < 2; ++j) {
      const unsigned short* p = &bs[(waveN * 32 + j * 16 + lrow) * 32 + koff];
      bf2[j].u[0] = *(const uint4*)(p);
      bf2[j].u[1] = *(const uint4*)(p + 16);
    }
#pragma unroll
    for (int i = 0; i < 4; ++i)
#pragma unroll
      for (int j = 0; j < 2; ++j)
        acc[i][j] = wmma_bf16(af[i], bf2[j], acc[i][j]);

#if !USE_ASYNC_COPY
    __syncthreads();
#endif
  }

#pragma unroll
  for (int j = 0; j < 2; ++j) {
    const int n = n0 + waveN * 32 + j * 16 + lrow;
    const float bv = bias[n];
#pragma unroll
    for (int i = 0; i < 4; ++i) {
      const int mb = m0 + waveM * 64 + i * 16 + half * 8;
#pragma unroll
      for (int r = 0; r < 8; ++r)
        C[(size_t)(mb + r) * N + n] = acc[i][j][r] + bv;
    }
  }
}

// ---------------------------------------------------------------------------
// Split fused qkv [B*T, 3072] f32 into bf16 buffers laid out for WMMA:
//   qb  [B, 32, T, 64]   (d contiguous  -> Q A-fragments)
//   kb  [B,  8, T, 64]   (d contiguous  -> K^T B-fragments)
//   vtb [B,  8, 64, T]   (t contiguous  -> V B-fragments for P*V)
// ---------------------------------------------------------------------------
__global__ void k_split_qkv(const float* __restrict__ qkv,
                            unsigned short* __restrict__ qb,
                            unsigned short* __restrict__ kb,
                            unsigned short* __restrict__ vtb) {
  const int total = 2 * 2048 * 3072;
  int i = blockIdx.x * blockDim.x + threadIdx.x;
  if (i >= total) return;
  int c  = i % 3072;
  int bt = i / 3072;
  int t  = bt & 2047;
  int b  = bt >> 11;
  unsigned short v = f32_to_bf16(qkv[i]);
  if (c < 2048) {
    int h = c >> 6, d = c & 63;
    qb[((((size_t)b * 32 + h) * 2048 + t) << 6) + d] = v;
  } else if (c < 2560) {
    int c2 = c - 2048, hk = c2 >> 6, d = c2 & 63;
    kb[((((size_t)b * 8 + hk) * 2048 + t) << 6) + d] = v;
  } else {
    int c2 = c - 2560, hk = c2 >> 6, d = c2 & 63;
    vtb[((((size_t)b * 8 + hk) * 64 + d) << 11) + t] = v;
  }
}

// ---------------------------------------------------------------------------
// Flash attention, causal, GQA (4 q-heads per kv-head).
// One wave (32 threads) per (b, h, 16-row Q tile); single-wave workgroup so
// s_barrier degenerates to S_NOP. Online softmax over 32-key tiles:
// 4 WMMA for S = Q*K^T, 4 WMMA for Y += P*V; P transposed via LDS.
// ---------------------------------------------------------------------------
__global__ __launch_bounds__(32) void k_flash_attn(
    const unsigned short* __restrict__ qb,
    const unsigned short* __restrict__ kb,
    const unsigned short* __restrict__ vtb,
    unsigned short* __restrict__ yb) {
  __shared__ unsigned short Pl[16 * 32];

  const int bid = blockIdx.x;
  const int qt  = bid & 127;
  const int h   = (bid >> 7) & 31;
  const int b   = bid >> 12;
  const int hk  = h >> 2;
  const int lane = threadIdx.x;
  const int lrow = lane & 15;
  const int half = lane >> 4;
  const int off8 = half * 8;
  const int q0 = qt * 16;

  // Q fragments (A layout): d 0..31 / 32..63, one 16-row tile.
  FragU a_lo, a_hi;
  {
    const unsigned short* qp =
        qb + ((((size_t)b * 32 + h) * 2048 + q0 + lrow) << 6);
    a_lo.u[0] = *(const uint4*)(qp + off8);
    a_lo.u[1] = *(const uint4*)(qp + off8 + 16);
    a_hi.u[0] = *(const uint4*)(qp + off8 + 32);
    a_hi.u[1] = *(const uint4*)(qp + off8 + 48);
  }

  const v8f vzero = {0.f, 0.f, 0.f, 0.f, 0.f, 0.f, 0.f, 0.f};
  v8f acc[4] = {vzero, vzero, vzero, vzero};   // Y 16x64 (4 n-chunks)
  float mrow[8], lsum[8];
#pragma unroll
  for (int r = 0; r < 8; ++r) { mrow[r] = -1e30f; lsum[r] = 0.f; }

  const unsigned short* kbase = kb  + (((size_t)b * 8 + hk) << 17); // *T*64
  const unsigned short* vbase = vtb + (((size_t)b * 8 + hk) << 17); // *64*T

  const int kend = q0 + 16;                    // keys needed (causal)
  for (int j0 = 0; j0 < kend; j0 += 32) {
    // ---- S = Q * K^T, two 16x16 tiles (keys j0..j0+15, j0+16..j0+31) ----
    v8f s0 = vzero, s1 = vzero;
    {
      const unsigned short* kp0 =
          kbase + (((size_t)(j0 + lrow)) << 6) + half * 8;
      const unsigned short* kp1 =
          kbase + (((size_t)(j0 + 16 + lrow)) << 6) + half * 8;
      FragU b0l, b0h, b1l, b1h;
      b0l.u[0] = *(const uint4*)(kp0);      b0l.u[1] = *(const uint4*)(kp0 + 16);
      b0h.u[0] = *(const uint4*)(kp0 + 32); b0h.u[1] = *(const uint4*)(kp0 + 48);
      b1l.u[0] = *(const uint4*)(kp1);      b1l.u[1] = *(const uint4*)(kp1 + 16);
      b1h.u[0] = *(const uint4*)(kp1 + 32); b1h.u[1] = *(const uint4*)(kp1 + 48);
      s0 = wmma_bf16(a_lo, b0l, s0);
      s0 = wmma_bf16(a_hi, b0h, s0);
      s1 = wmma_bf16(a_lo, b1l, s1);
      s1 = wmma_bf16(a_hi, b1h, s1);
    }
    if (j0 + 32 < kend) {                      // speculative prefetch next tile
      __builtin_prefetch(kbase + ((size_t)(j0 + 32 + lrow) << 6), 0, 0);
    }

    // ---- scale, causal mask, online softmax (C layout: m=r+8*half, n=lrow) --
    float fac[8];
#pragma unroll
    for (int r = 0; r < 8; ++r) {
      const int qi = q0 + r + 8 * half;
      const int kj = j0 + lrow;
      float v0 = (kj      <= qi) ? s0[r] * 0.125f : -1e30f;
      float v1 = (kj + 16 <= qi) ? s1[r] * 0.125f : -1e30f;
      float mx = fmaxf(v0, v1);
      mx = fmaxf(mx, __shfl_xor(mx, 1, 16));
      mx = fmaxf(mx, __shfl_xor(mx, 2, 16));
      mx = fmaxf(mx, __shfl_xor(mx, 4, 16));
      mx = fmaxf(mx, __shfl_xor(mx, 8, 16));
      const float mnew = fmaxf(mrow[r], mx);
      const float f  = __expf(mrow[r] - mnew);
      const float p0 = __expf(v0 - mnew);
      const float p1 = __expf(v1 - mnew);
      float sum = p0 + p1;
      sum += __shfl_xor(sum, 1, 16);
      sum += __shfl_xor(sum, 2, 16);
      sum += __shfl_xor(sum, 4, 16);
      sum += __shfl_xor(sum, 8, 16);
      lsum[r] = lsum[r] * f + sum;
      mrow[r] = mnew;
      fac[r]  = f;
      Pl[(r + 8 * half) * 32 + lrow]      = f32_to_bf16(p0);
      Pl[(r + 8 * half) * 32 + lrow + 16] = f32_to_bf16(p1);
    }
#pragma unroll
    for (int nt = 0; nt < 4; ++nt)
#pragma unroll
      for (int r = 0; r < 8; ++r) acc[nt][r] *= fac[r];

    __syncthreads();   // make lane-crossed P tile visible (wave-level)

    // ---- P (A layout, 16x32) from LDS ----
    FragU pa;
    {
      const unsigned short* pp = &Pl[lrow * 32];
      pa.u[0] = *(const uint4*)(pp + off8);
      pa.u[1] = *(const uint4*)(pp + off8 + 16);
    }

    // ---- Y += P * V  (V^T buffer: lane n = d, K = key index, contiguous) ----
#pragma unroll
    for (int nt = 0; nt < 4; ++nt) {
      const unsigned short* vp =
          vbase + (((size_t)(nt * 16 + lrow)) << 11) + j0 + half * 8;
      FragU vb;
      vb.u[0] = *(const uint4*)(vp);
      vb.u[1] = *(const uint4*)(vp + 16);
      acc[nt] = wmma_bf16(pa, vb, acc[nt]);
    }
    __syncthreads();   // Pl reused next iteration
  }

  // ---- normalize rows and store y (bf16, [B*T, 2048], col = h*64 + d) ----
  float rl[8];
#pragma unroll
  for (int r = 0; r < 8; ++r) rl[r] = 1.0f / lsum[r];
#pragma unroll
  for (int nt = 0; nt < 4; ++nt) {
    const int d = nt * 16 + lrow;
#pragma unroll
    for (int r = 0; r < 8; ++r) {
      const int m = q0 + r + 8 * half;
      yb[(((size_t)b * 2048 + m) << 11) + h * 64 + d] =
          f32_to_bf16(acc[nt][r] * rl[r]);
    }
  }
}

// ---------------------------------------------------------------------------
// Host-side launch
// ---------------------------------------------------------------------------
extern "C" void kernel_launch(void* const* d_in, const int* in_sizes, int n_in,
                              void* d_out, int out_size, void* d_ws, size_t ws_size,
                              hipStream_t stream) {
  const float* x      = (const float*)d_in[0];
  const float* w_attn = (const float*)d_in[1];
  const float* b_attn = (const float*)d_in[2];
  const float* w_proj = (const float*)d_in[3];
  const float* b_proj = (const float*)d_in[4];

  const int B = 2, T = 2048, C = 2048, QKVW = 3072, NH = 32, NKV = 8, HS = 64;
  const int M = B * T;                    // 4096

  char* wsp = (char*)d_ws;
  size_t off = 0;
  auto take = [&](size_t bytes) -> char* {
    char* p = wsp + off;
    off += (bytes + 255) & ~(size_t)255;
    return p;
  };
  unsigned short* xb  = (unsigned short*)take((size_t)M * C * 2);
  unsigned short* wab = (unsigned short*)take((size_t)QKVW * C * 2);
  unsigned short* wpb = (unsigned short*)take((size_t)C * C * 2);
  float*          qkv = (float*)take((size_t)M * QKVW * 4);
  unsigned short* qb_ = (unsigned short*)take((size_t)B * NH  * T * HS * 2);
  unsigned short* kb_ = (unsigned short*)take((size_t)B * NKV * T * HS * 2);
  unsigned short* vtb = (unsigned short*)take((size_t)B * NKV * HS * T * 2);
  unsigned short* yb  = (unsigned short*)take((size_t)M * C * 2);

  const int n1 = M * C;
  k_cast_bf16<<<(n1 + 255) / 256, 256, 0, stream>>>(x, xb, n1);
  const int n2 = QKVW * C;
  k_cast_bf16<<<(n2 + 255) / 256, 256, 0, stream>>>(w_attn, wab, n2);
  const int n3 = C * C;
  k_cast_bf16<<<(n3 + 255) / 256, 256, 0, stream>>>(w_proj, wpb, n3);

  // qkv = x @ w_attn^T + b_attn
  k_gemm_bf16<<<dim3(QKVW / 128, M / 128), 256, 0, stream>>>(
      xb, wab, b_attn, qkv, M, QKVW, C);

  const int nq = M * QKVW;
  k_split_qkv<<<(nq + 255) / 256, 256, 0, stream>>>(qkv, qb_, kb_, vtb);

  // flash attention: one wave per (b, h, 16-row q tile)
  k_flash_attn<<<B * NH * (T / 16), 32, 0, stream>>>(qb_, kb_, vtb, yb);

  // out = y @ w_proj^T + b_proj
  k_gemm_bf16<<<dim3(C / 128, M / 128), 256, 0, stream>>>(
      yb, wpb, b_proj, (float*)d_out, M, C, C);
}
